// _HLRTMLP_11407433138400
// MI455X (gfx1250) — matrix operand order repaired and checked by reading, not compiled, not run on hardware
//
#include <hip/hip_runtime.h>
#include <cstdint>
#include <cstddef>

// ---------------- types ----------------
typedef __attribute__((ext_vector_type(16))) __bf16 v16bf;
typedef __attribute__((ext_vector_type(8)))  float  v8f;

union FragU { v16bf v; uint4 q[2]; };

__device__ __forceinline__ unsigned short f2bf(float f) {
    union { float f; unsigned u; } v; v.f = f;
    unsigned r = v.u + 0x7FFFu + ((v.u >> 16) & 1u);
    return (unsigned short)(r >> 16);
}
__device__ __forceinline__ float bf2f(unsigned short h) {
    union { unsigned u; float f; } v; v.u = ((unsigned)h) << 16; return v.f;
}
__device__ __forceinline__ float sigm(float x) { return 1.f / (1.f + expf(-x)); }

// ---------------- WMMA tile: D(16x16) += A(16xK) * B(Kx16), A row-major LDS, B pre-transposed [N][K] ----------------
template<int KLEN>
__device__ __forceinline__ v8f wmma_loop(const unsigned short* A, int lda,
                                         const unsigned short* Bt, int ldb, int lane) {
    v8f acc;
#pragma unroll
    for (int i = 0; i < 8; ++i) acc[i] = 0.f;
    const int hf = lane >> 4;
    const unsigned short* aRow = A + (lane & 15) * lda;
    const unsigned short* bRow = Bt + (lane & 15) * ldb;
#pragma unroll
    for (int k0 = 0; k0 < KLEN; k0 += 32) {
        FragU fa, fb;
        // A 16x32 bf16: lanes0-15 K=[0..7]+[16..23], lanes16-31 K=[8..15]+[24..31]
        fa.q[0] = *(const uint4*)(aRow + k0 + hf * 8);
        fa.q[1] = *(const uint4*)(aRow + k0 + hf * 8 + 16);
        // B 32x16 bf16: lanes0-15 K=[0..7]+[16..23], lanes16-31 K=[8..15]+[24..31], column N = lane&15
        fb.q[0] = *(const uint4*)(bRow + k0 + hf * 8);
        fb.q[1] = *(const uint4*)(bRow + k0 + hf * 8 + 16);
        acc = __builtin_amdgcn_wmma_f32_16x16x32_bf16(false, fa.v, false, fb.v,
                                                      (short)0, acc, false, false);
    }
    return acc;
}

// ---------------- prep kernel: bf16 transposed / LN-folded weights ----------------
struct PrepP {
    const float *w1, *w2, *w3;                       // gate1.w[512,256], gate2.w[256,512], backbone.w[512,256]
    const float *ss, *sb, *wsafe, *bsafe;            // safe ln_s/ln_b[256], l1.w[256,128], l1.b[128]
    const float *as_, *ab_, *aw, *abias;             // a_head ln_s/ln_b[272], l1.w[272,68], l1.b[68]
    const float *bs_, *bb_, *bw, *bbias;             // b_head
    const float *ns_, *nb_, *nw, *nbias;             // nl_head
    unsigned short *W1t, *W2t, *W3t, *Wst, *Wabnt;   // [256,512] [512,256] [256,512] [128,256] [208,288]
    float *BS, *BABN;                                // [128] [208]
};

__global__ __launch_bounds__(256) void prep_kernel(PrepP q) {
    int g = blockIdx.x * blockDim.x + threadIdx.x;
    int st = gridDim.x * blockDim.x;
    for (int i = g; i < 256 * 512; i += st) { int n = i >> 9, k = i & 511; q.W1t[i] = f2bf(q.w1[k * 256 + n]); }
    for (int i = g; i < 512 * 256; i += st) { int n = i >> 8, k = i & 255; q.W2t[i] = f2bf(q.w2[k * 512 + n]); }
    for (int i = g; i < 256 * 512; i += st) { int n = i >> 9, k = i & 511; q.W3t[i] = f2bf(q.w3[k * 256 + n]); }
    for (int i = g; i < 128 * 256; i += st) { int n = i >> 8, k = i & 255; q.Wst[i] = f2bf(q.ss[k] * q.wsafe[k * 128 + n]); }
    for (int i = g; i < 208 * 288; i += st) {
        int n = i / 288, k = i % 288;
        float v = 0.f;
        if (k < 272 && n < 204) {
            const float *s, *w; int j;
            if (n < 68)       { s = q.as_; w = q.aw; j = n; }
            else if (n < 136) { s = q.bs_; w = q.bw; j = n - 68; }
            else              { s = q.ns_; w = q.nw; j = n - 136; }
            v = s[k] * w[k * 68 + j];
        }
        q.Wabnt[i] = f2bf(v);
    }
    for (int i = g; i < 128; i += st) {
        float acc = q.bsafe[i];
        for (int k = 0; k < 256; ++k) acc += q.sb[k] * q.wsafe[k * 128 + i];
        q.BS[i] = acc;
    }
    for (int i = g; i < 208; i += st) {
        float acc = 0.f;
        if (i < 204) {
            const float *lb, *w, *bi; int j;
            if (i < 68)       { lb = q.ab_; w = q.aw; bi = q.abias; j = i; }
            else if (i < 136) { lb = q.bb_; w = q.bw; bi = q.bbias; j = i - 68; }
            else              { lb = q.nb_; w = q.nw; bi = q.nbias; j = i - 136; }
            acc = bi[j];
            for (int k = 0; k < 272; ++k) acc += lb[k] * w[k * 68 + j];
        }
        q.BABN[i] = acc;
    }
}

// ---------------- main fused kernel ----------------
struct Params {
    const float *x;
    const float *b1, *b2, *b3;            // gate1.b[256], gate2.b[512], backbone.b[256]
    const float *w_base, *b_base;         // [256,1],[1]
    const float *w_safe2, *b_safe2;       // [128,1],[1]
    const float *w_sum1, *b_sum1;         // [4,16],[16]
    const float *w_sum2, *b_sum2;         // [16,16],[16]
    const float *w_a2, *b_a2;             // [68,8],[8]
    const float *w_b2, *b_b2;             // [68,1],[1]
    const float *w_n2, *b_n2;             // [68,1],[1]
    const float *nlg_s, *nlg_b, *nlg_w1, *nlg_b1, *nlg_w2, *nlg_b2;  // [259]x2,[259,64],[64],[64,1],[1]
    const float *cg_s, *cg_b, *cg_w1, *cg_b1, *cg_w2, *cg_b2;        // [262]x2,[262,65],[65],[65,1],[1]
    const unsigned short *W1t, *W2t, *W3t, *Wst, *Wabnt;
    const float *BS, *BABN;
    float *out;
};

__global__ __launch_bounds__(256) void fused_kernel(Params p) {
    const int tid  = threadIdx.x;
    const int wid  = tid >> 5, lane = tid & 31;
    const int r    = tid >> 4, sub  = tid & 15;
    const int row0 = blockIdx.x * 16;

    __shared__ __align__(16) unsigned short U1[16 * 528];   // x bf16 -> g bf16
    __shared__ __align__(16) unsigned short U2[16 * 272];   // h1 bf16
    __shared__ __align__(16) unsigned short U3[16 * 528];   // topk scratch -> xg bf16 -> xhat[16*304]
    __shared__ __align__(16) unsigned short sHb[16 * 272];  // h_base bf16
    __shared__ __align__(16) unsigned short sHid[16 * 208]; // head hidden bf16
    __shared__ float sScal[16][64];
    __shared__ float sRedA[16][16], sRedB[16][16], sRedC[16][16];

    // phase 1: x tile -> bf16 LDS
    for (int i = tid; i < 16 * 512; i += 256) {
        int rr = i >> 9, c = i & 511;
        U1[rr * 528 + c] = f2bf(p.x[(size_t)(row0 + rr) * 512 + c]);
    }
    __syncthreads();

    // GEMM1: h1 = relu(x @ W1 + b1)   [16,512]x[512,256]
    for (int t = 0; t < 2; ++t) {
        int nb = (wid * 2 + t) * 16;
        v8f acc = wmma_loop<512>(U1, 528, p.W1t + nb * 512, 512, lane);
        int n = nb + (lane & 15), hf = lane >> 4;
        float bb = p.b1[n];
#pragma unroll
        for (int v = 0; v < 8; ++v)
            U2[(v + 8 * hf) * 272 + n] = f2bf(fmaxf(acc[v] + bb, 0.f));
    }
    __syncthreads();

    // GEMM2: g = sigmoid(h1 @ W2 + b2)  [16,256]x[256,512]
    for (int t = 0; t < 4; ++t) {
        int nb = (wid * 4 + t) * 16;
        v8f acc = wmma_loop<256>(U2, 272, p.W2t + nb * 256, 256, lane);
        int n = nb + (lane & 15), hf = lane >> 4;
        float bb = p.b2[n];
#pragma unroll
        for (int v = 0; v < 8; ++v)
            U1[(v + 8 * hf) * 528 + n] = f2bf(sigm(acc[v] + bb));
    }
    __syncthreads();

    // gate stats + per-thread top-8 (packed key = bf16bits<<16 | col)
    {
        unsigned* cand = (unsigned*)U3;
        float s = 0.f, ssq = 0.f, mx = -1e30f;
        unsigned top[8];
#pragma unroll
        for (int j = 0; j < 8; ++j) top[j] = 0u;
        for (int c = sub; c < 512; c += 16) {
            unsigned short hb = U1[r * 528 + c];
            float gv = bf2f(hb);
            s += gv; ssq += gv * gv; mx = fmaxf(mx, gv);
            unsigned key = ((unsigned)hb << 16) | (unsigned)c;
            if (key > top[7]) {
                int pos = 7;
                while (pos > 0 && key > top[pos - 1]) { top[pos] = top[pos - 1]; --pos; }
                top[pos] = key;
            }
        }
        sRedA[r][sub] = s; sRedB[r][sub] = ssq; sRedC[r][sub] = mx;
#pragma unroll
        for (int j = 0; j < 8; ++j) cand[(r << 7) + sub * 8 + j] = top[j];
    }
    __syncthreads();
    if (sub == 0) {
        const unsigned* cand = (const unsigned*)U3;
        float s = 0.f, ssq = 0.f, mx = -1e30f;
        for (int i = 0; i < 16; ++i) { s += sRedA[r][i]; ssq += sRedB[r][i]; mx = fmaxf(mx, sRedC[r][i]); }
        float gm = s * (1.f / 512.f);
        sScal[r][16] = gm;
        sScal[r][17] = mx;
        sScal[r][18] = sqrtf(fmaxf(ssq * (1.f / 512.f) - gm * gm, 0.f));
        // merge top-8 of 128 candidates
        float tg[8], tx[8];
        unsigned prev = 0xFFFFFFFFu;
        for (int j = 0; j < 8; ++j) {
            unsigned best = 0u;
            for (int i = 0; i < 128; ++i) {
                unsigned k = cand[(r << 7) + i];
                if (k < prev && k > best) best = k;
            }
            prev = best;
            tg[j] = bf2f((unsigned short)(best >> 16));
            int idx = (int)(best & 0xFFFFu);
            tx[j] = p.x[(size_t)(row0 + r) * 512 + idx];
            sScal[r][20 + j] = tx[j];
        }
        float tsum = 0.f, wmax = -1e30f;
        for (int j = 0; j < 8; ++j) { tsum += tg[j]; wmax = fmaxf(wmax, tx[j]); }
        float inv = 1.f / (tsum + 1e-6f);
        float wmean = 0.f;
        for (int j = 0; j < 8; ++j) wmean += tg[j] * inv * tx[j];
        float wvar = 0.f;
        for (int j = 0; j < 8; ++j) { float d = tx[j] - wmean; wvar += tg[j] * inv * d * d; }
        float wstd = sqrtf(wvar + 1e-6f);
        float gmean_t = tsum * 0.125f;
        float gconc = tg[0] / (gmean_t + 1e-6f);
        float raw[4] = { wmean, wstd, wmax, gconc };
        float h1s[16];
        for (int j = 0; j < 16; ++j) {
            float v = p.b_sum1[j];
            for (int i = 0; i < 4; ++i) v += raw[i] * p.w_sum1[i * 16 + j];
            h1s[j] = fmaxf(v, 0.f);
        }
        float zn = 0.f;
        for (int j = 0; j < 16; ++j) {
            float v = p.b_sum2[j];
            for (int i = 0; i < 16; ++i) v += h1s[i] * p.w_sum2[i * 16 + j];
            sScal[r][j] = v; zn += v * v;
        }
        sScal[r][28] = sqrtf(zn);  // z_norm
    }
    __syncthreads();

    // xg = x * g -> U3 bf16
    for (int i = tid; i < 16 * 512; i += 256) {
        int rr = i >> 9, c = i & 511;
        float xv = p.x[(size_t)(row0 + rr) * 512 + c];
        U3[rr * 528 + c] = f2bf(xv * bf2f(U1[rr * 528 + c]));
    }
    __syncthreads();

    // GEMM3: h_base = relu(xg @ Wb + b3)
    for (int t = 0; t < 2; ++t) {
        int nb = (wid * 2 + t) * 16;
        v8f acc = wmma_loop<512>(U3, 528, p.W3t + nb * 512, 512, lane);
        int n = nb + (lane & 15), hf = lane >> 4;
        float bb = p.b3[n];
#pragma unroll
        for (int v = 0; v < 8; ++v)
            sHb[(v + 8 * hf) * 272 + n] = f2bf(fmaxf(acc[v] + bb, 0.f));
    }
    __syncthreads();

    // y_base dot + LN stats of h_base (fused pass)
    {
        float pv = 0.f, s = 0.f, ssq = 0.f;
        for (int c = sub; c < 256; c += 16) {
            float h = bf2f(sHb[r * 272 + c]);
            pv += h * p.w_base[c]; s += h; ssq += h * h;
        }
        sRedA[r][sub] = pv; sRedB[r][sub] = s; sRedC[r][sub] = ssq;
    }
    __syncthreads();
    if (sub == 0) {
        float pv = 0.f, s = 0.f, ssq = 0.f;
        for (int i = 0; i < 16; ++i) { pv += sRedA[r][i]; s += sRedB[r][i]; ssq += sRedC[r][i]; }
        sScal[r][19] = pv + p.b_base[0];
        float mu = s * (1.f / 256.f);
        sScal[r][30] = mu;
        sScal[r][31] = rsqrtf(fmaxf(ssq * (1.f / 256.f) - mu * mu, 0.f) + 1e-5f);
    }
    __syncthreads();

    // safe head: xhat(h_base) -> U3 (stride 304), WMMA 256->128 with folded LN, then dot 128->1
    for (int i = tid; i < 16 * 304; i += 256) {
        int rr = i / 304, c = i % 304;
        float v = 0.f;
        if (c < 256) v = (bf2f(sHb[rr * 272 + c]) - sScal[rr][30]) * sScal[rr][31];
        U3[rr * 304 + c] = f2bf(v);
    }
    __syncthreads();
    {
        int nb = wid * 16;
        v8f acc = wmma_loop<256>(U3, 304, p.Wst + nb * 256, 256, lane);
        int n = nb + (lane & 15), hf = lane >> 4;
        float bb = p.BS[n];
#pragma unroll
        for (int v = 0; v < 8; ++v)
            sHid[(v + 8 * hf) * 208 + n] = f2bf(fmaxf(acc[v] + bb, 0.f));
    }
    __syncthreads();
    {
        float pv = 0.f;
        for (int c = sub; c < 128; c += 16) pv += bf2f(sHid[r * 208 + c]) * p.w_safe2[c];
        sRedA[r][sub] = pv;
    }
    __syncthreads();
    if (sub == 0) {
        float pv = 0.f;
        for (int i = 0; i < 16; ++i) pv += sRedA[r][i];
        sScal[r][29] = pv + p.b_safe2[0];  // delta_safe
    }
    __syncthreads();

    // lar = [h_base, z_top] LN stats
    {
        float s = 0.f, ssq = 0.f;
        for (int c = sub; c < 272; c += 16) {
            float v = (c < 256) ? bf2f(sHb[r * 272 + c]) : sScal[r][c - 256];
            s += v; ssq += v * v;
        }
        sRedA[r][sub] = s; sRedB[r][sub] = ssq;
    }
    __syncthreads();
    if (sub == 0) {
        float s = 0.f, ssq = 0.f;
        for (int i = 0; i < 16; ++i) { s += sRedA[r][i]; ssq += sRedB[r][i]; }
        float mu = s * (1.f / 272.f);
        sScal[r][30] = mu;
        sScal[r][31] = rsqrtf(fmaxf(ssq * (1.f / 272.f) - mu * mu, 0.f) + 1e-5f);
    }
    __syncthreads();
    for (int i = tid; i < 16 * 304; i += 256) {
        int rr = i / 304, c = i % 304;
        float v = 0.f;
        if (c < 272) {
            float lv = (c < 256) ? bf2f(sHb[rr * 272 + c]) : sScal[rr][c - 256];
            v = (lv - sScal[rr][30]) * sScal[rr][31];
        }
        U3[rr * 304 + c] = f2bf(v);
    }
    __syncthreads();

    // fused a/b/nl l1: WMMA [16,288]x[288,208] with LN folded
    for (int nt = wid; nt < 13; nt += 8) {
        int nb = nt * 16;
        v8f acc = wmma_loop<288>(U3, 304, p.Wabnt + nb * 288, 288, lane);
        int n = nb + (lane & 15), hf = lane >> 4;
        float bb = p.BABN[n];
#pragma unroll
        for (int v = 0; v < 8; ++v)
            sHid[(v + 8 * hf) * 208 + n] = f2bf(fmaxf(acc[v] + bb, 0.f));
    }
    __syncthreads();

    // a/b/nl l2
    if (sub < 8) {
        float v = p.b_a2[sub];
        for (int c = 0; c < 68; ++c) v += bf2f(sHid[r * 208 + c]) * p.w_a2[c * 8 + sub];
        sScal[r][36 + sub] = 0.2f * tanhf(v);
    } else if (sub == 8) {
        float v = p.b_b2[0];
        for (int c = 0; c < 68; ++c) v += bf2f(sHid[r * 208 + 68 + c]) * p.w_b2[c];
        sScal[r][33] = 0.08f * tanhf(v);
    } else if (sub == 9) {
        float v = p.b_n2[0];
        for (int c = 0; c < 68; ++c) v += bf2f(sHid[r * 208 + 136 + c]) * p.w_n2[c];
        sScal[r][45] = 0.1f * tanhf(v);  // delta_nl
    }
    __syncthreads();
    if (sub == 0) {
        float dl = sScal[r][33];
        for (int j = 0; j < 8; ++j) dl += sScal[r][36 + j] * sScal[r][20 + j];
        sScal[r][44] = dl;  // delta_lin
    }
    __syncthreads();

    // nl_gate: u = [h_base, z_norm, |dlin|, |dnl|] (259), LN + 259->64 relu -> 64->1 sigmoid
    {
        float s = 0.f, ssq = 0.f;
        for (int c = sub; c < 259; c += 16) {
            float v = (c < 256) ? bf2f(sHb[r * 272 + c])
                    : (c == 256) ? sScal[r][28]
                    : (c == 257) ? fabsf(sScal[r][44]) : fabsf(sScal[r][45]);
            s += v; ssq += v * v;
        }
        sRedA[r][sub] = s; sRedB[r][sub] = ssq;
    }
    __syncthreads();
    if (sub == 0) {
        float s = 0.f, ssq = 0.f;
        for (int i = 0; i < 16; ++i) { s += sRedA[r][i]; ssq += sRedB[r][i]; }
        float mu = s * (1.f / 259.f);
        sScal[r][30] = mu;
        sScal[r][31] = rsqrtf(fmaxf(ssq * (1.f / 259.f) - mu * mu, 0.f) + 1e-5f);
    }
    __syncthreads();
    for (int c = sub; c < 259; c += 16) {
        float v = (c < 256) ? bf2f(sHb[r * 272 + c])
                : (c == 256) ? sScal[r][28]
                : (c == 257) ? fabsf(sScal[r][44]) : fabsf(sScal[r][45]);
        float xh = (v - sScal[r][30]) * sScal[r][31];
        U3[r * 304 + c] = f2bf(xh * p.nlg_s[c] + p.nlg_b[c]);
    }
    __syncthreads();
    for (int j = sub; j < 64; j += 16) {
        float v = p.nlg_b1[j];
        for (int c = 0; c < 259; ++c) v += bf2f(U3[r * 304 + c]) * p.nlg_w1[c * 64 + j];
        sHid[r * 208 + j] = f2bf(fmaxf(v, 0.f));
    }
    __syncthreads();
    {
        float pv = 0.f;
        for (int j = sub; j < 64; j += 16) pv += bf2f(sHid[r * 208 + j]) * p.nlg_w2[j];
        sRedA[r][sub] = pv;
    }
    __syncthreads();
    if (sub == 0) {
        float pv = 0.f;
        for (int i = 0; i < 16; ++i) pv += sRedA[r][i];
        float eta = sigm(pv + p.nlg_b2[0]);
        sScal[r][47] = sScal[r][44] + eta * sScal[r][45];  // spec_total
    }
    __syncthreads();

    // conf_gate: u = [h_base, y_base, gmean, gmax, gstd, |dsafe|, |spec|] (262), LN + 262->65 relu -> 65->1 sigmoid
    {
        float s = 0.f, ssq = 0.f;
        for (int c = sub; c < 262; c += 16) {
            float v = (c < 256) ? bf2f(sHb[r * 272 + c])
                    : (c == 256) ? sScal[r][19]
                    : (c == 257) ? sScal[r][16]
                    : (c == 258) ? sScal[r][17]
                    : (c == 259) ? sScal[r][18]
                    : (c == 260) ? fabsf(sScal[r][29]) : fabsf(sScal[r][47]);
            s += v; ssq += v * v;
        }
        sRedA[r][sub] = s; sRedB[r][sub] = ssq;
    }
    __syncthreads();
    if (sub == 0) {
        float s = 0.f, ssq = 0.f;
        for (int i = 0; i < 16; ++i) { s += sRedA[r][i]; ssq += sRedB[r][i]; }
        float mu = s * (1.f / 262.f);
        sScal[r][30] = mu;
        sScal[r][31] = rsqrtf(fmaxf(ssq * (1.f / 262.f) - mu * mu, 0.f) + 1e-5f);
    }
    __syncthreads();
    for (int c = sub; c < 262; c += 16) {
        float v = (c < 256) ? bf2f(sHb[r * 272 + c])
                : (c == 256) ? sScal[r][19]
                : (c == 257) ? sScal[r][16]
                : (c == 258) ? sScal[r][17]
                : (c == 259) ? sScal[r][18]
                : (c == 260) ? fabsf(sScal[r][29]) : fabsf(sScal[r][47]);
        float xh = (v - sScal[r][30]) * sScal[r][31];
        U3[r * 304 + c] = f2bf(xh * p.cg_s[c] + p.cg_b[c]);
    }
    __syncthreads();
    for (int j = sub; j < 65; j += 16) {
        float v = p.cg_b1[j];
        for (int c = 0; c < 262; ++c) v += bf2f(U3[r * 304 + c]) * p.cg_w1[c * 65 + j];
        sHid[r * 208 + j] = f2bf(fmaxf(v, 0.f));
    }
    __syncthreads();
    {
        float pv = 0.f;
        for (int j = sub; j < 65; j += 16) pv += bf2f(sHid[r * 208 + j]) * p.cg_w2[j];
        sRedA[r][sub] = pv;
    }
    __syncthreads();
    if (sub == 0) {
        float pv = 0.f;
        for (int i = 0; i < 16; ++i) pv += sRedA[r][i];
        float gamma = sigm(pv + p.cg_b2[0]);
        p.out[row0 + r] = sScal[r][19] + sScal[r][29] + gamma * sScal[r][47];
    }
}

// ---------------- launch ----------------
extern "C" void kernel_launch(void* const* d_in, const int* in_sizes, int n_in,
                              void* d_out, int out_size, void* d_ws, size_t ws_size,
                              hipStream_t stream) {
    // input flattening (setup_inputs dict insertion order)
    const float* x        = (const float*)d_in[0];
    // d_in[1] = x_cat (unused)
    const float* g1w = (const float*)d_in[2],  *g1b = (const float*)d_in[3];
    const float* g2w = (const float*)d_in[4],  *g2b = (const float*)d_in[5];
    const float* bbw = (const float*)d_in[6],  *bbb = (const float*)d_in[7];
    const float* bhw = (const float*)d_in[8],  *bhb = (const float*)d_in[9];
    const float* sf_s = (const float*)d_in[10], *sf_b = (const float*)d_in[11];
    const float* sf_w1 = (const float*)d_in[12], *sf_b1 = (const float*)d_in[13];
    const float* sf_w2 = (const float*)d_in[14], *sf_b2 = (const float*)d_in[15];
    const float* s1w = (const float*)d_in[16], *s1b = (const float*)d_in[17];
    const float* s2w = (const float*)d_in[18], *s2b = (const float*)d_in[19];
    const float* a_s = (const float*)d_in[20], *a_b = (const float*)d_in[21];
    const float* a_w1 = (const float*)d_in[22], *a_b1 = (const float*)d_in[23];
    const float* a_w2 = (const float*)d_in[24], *a_b2 = (const float*)d_in[25];
    const float* b_s = (const float*)d_in[26], *b_b = (const float*)d_in[27];
    const float* b_w1 = (const float*)d_in[28], *b_b1 = (const float*)d_in[29];
    const float* b_w2 = (const float*)d_in[30], *b_b2 = (const float*)d_in[31];
    const float* n_s = (const float*)d_in[32], *n_b = (const float*)d_in[33];
    const float* n_w1 = (const float*)d_in[34], *n_b1 = (const float*)d_in[35];
    const float* n_w2 = (const float*)d_in[36], *n_b2 = (const float*)d_in[37];
    const float* nlg_s = (const float*)d_in[38], *nlg_b = (const float*)d_in[39];
    const float* nlg_w1 = (const float*)d_in[40], *nlg_b1 = (const float*)d_in[41];
    const float* nlg_w2 = (const float*)d_in[42], *nlg_b2 = (const float*)d_in[43];
    const float* cg_s = (const float*)d_in[44], *cg_b = (const float*)d_in[45];
    const float* cg_w1 = (const float*)d_in[46], *cg_b1 = (const float*)d_in[47];
    const float* cg_w2 = (const float*)d_in[48], *cg_b2 = (const float*)d_in[49];

    // workspace layout (bytes)
    unsigned char* ws = (unsigned char*)d_ws;
    const size_t oW1t = 0;
    const size_t oW2t = oW1t + 256 * 512 * 2;     // 262144
    const size_t oW3t = oW2t + 512 * 256 * 2;     // 524288
    const size_t oWst = oW3t + 256 * 512 * 2;     // 786432
    const size_t oWab = oWst + 128 * 256 * 2;     // 851968
    const size_t oBS  = oWab + 208 * 288 * 2;     // 971776
    const size_t oBA  = oBS + 128 * 4;            // 972288
    const size_t need = oBA + 208 * 4;            // 973120
    if (ws_size < need) return;

    unsigned short* W1t = (unsigned short*)(ws + oW1t);
    unsigned short* W2t = (unsigned short*)(ws + oW2t);
    unsigned short* W3t = (unsigned short*)(ws + oW3t);
    unsigned short* Wst = (unsigned short*)(ws + oWst);
    unsigned short* Wab = (unsigned short*)(ws + oWab);
    float* BS = (float*)(ws + oBS);
    float* BA = (float*)(ws + oBA);

    PrepP q;
    q.w1 = g1w; q.w2 = g2w; q.w3 = bbw;
    q.ss = sf_s; q.sb = sf_b; q.wsafe = sf_w1; q.bsafe = sf_b1;
    q.as_ = a_s; q.ab_ = a_b; q.aw = a_w1; q.abias = a_b1;
    q.bs_ = b_s; q.bb_ = b_b; q.bw = b_w1; q.bbias = b_b1;
    q.ns_ = n_s; q.nb_ = n_b; q.nw = n_w1; q.nbias = n_b1;
    q.W1t = W1t; q.W2t = W2t; q.W3t = W3t; q.Wst = Wst; q.Wabnt = Wab;
    q.BS = BS; q.BABN = BA;
    prep_kernel<<<512, 256, 0, stream>>>(q);

    Params p;
    p.x = x;
    p.b1 = g1b; p.b2 = g2b; p.b3 = bbb;
    p.w_base = bhw; p.b_base = bhb;
    p.w_safe2 = sf_w2; p.b_safe2 = sf_b2;
    p.w_sum1 = s1w; p.b_sum1 = s1b; p.w_sum2 = s2w; p.b_sum2 = s2b;
    p.w_a2 = a_w2; p.b_a2 = a_b2;
    p.w_b2 = b_w2; p.b_b2 = b_b2;
    p.w_n2 = n_w2; p.b_n2 = n_b2;
    p.nlg_s = nlg_s; p.nlg_b = nlg_b; p.nlg_w1 = nlg_w1; p.nlg_b1 = nlg_b1;
    p.nlg_w2 = nlg_w2; p.nlg_b2 = nlg_b2;
    p.cg_s = cg_s; p.cg_b = cg_b; p.cg_w1 = cg_w1; p.cg_b1 = cg_b1;
    p.cg_w2 = cg_w2; p.cg_b2 = cg_b2;
    p.W1t = W1t; p.W2t = W2t; p.W3t = W3t; p.Wst = Wst; p.Wabnt = Wab;
    p.BS = BS; p.BABN = BA;
    p.out = (float*)d_out;
    fused_kernel<<<131072 / 16, 256, 0, stream>>>(p);
}
